// SimpleTransformerLayer_9079560864061
// MI455X (gfx1250) — matrix operand order repaired and checked by reading, not compiled, not run on hardware
//
#include <hip/hip_runtime.h>
#include <math.h>

// ---------- types ----------
typedef __bf16        v16bf __attribute__((ext_vector_type(16)));
typedef float         v8f   __attribute__((ext_vector_type(8)));
typedef unsigned int  v4u   __attribute__((ext_vector_type(4)));

union FragBF { v4u u[2]; v16bf v; };

#define WMMA_BF16(a, b, c) \
  __builtin_amdgcn_wmma_f32_16x16x32_bf16(false, (a), false, (b), (short)0, (c), false, false)

// async 16B global->LDS copy (GVS mode: sgpr base + vgpr byte offset), ASYNCcnt-tracked
#define ASYNC_CP_B128(ldsoff_u32, voff_u32, sbase_ptr)                        \
  asm volatile("global_load_async_to_lds_b128 %0, %1, %2"                     \
               :: "v"(ldsoff_u32), "v"(voff_u32), "s"(sbase_ptr) : "memory")
#define WAIT_ASYNC0() asm volatile("s_wait_asynccnt 0" ::: "memory")

static __device__ __forceinline__ unsigned short f2bf(float f) {
  unsigned int u = __float_as_uint(f);
  u += 0x7FFFu + ((u >> 16) & 1u);          // round-to-nearest-even
  return (unsigned short)(u >> 16);
}

static __device__ __forceinline__ float gelu_exact(float x) {
  return 0.5f * x * (1.0f + erff(x * 0.70710678118654752f));
}

// ---------- model constants ----------
#define BATCH 4
#define SEQ   2048
#define DM    1024
#define NH    16
#define HD    64
#define DFF   4096
#define MROWS (BATCH * SEQ)   // 8192

// LDS tile geometry: 128 rows x 32 bf16, padded to 40-element rows (80B stride:
// 20-bank stride -> the 16 lanes of a b128 fragment read hit 16 distinct banks)
#define TROW   40
#define THALF  (128 * TROW)       // halfs per tile (5120)
#define TBYTES (THALF * 2)        // 10240 bytes per tile
#define CHUNK1 (64 * TROW * 2)    // byte offset of second copy-chunk (rows +64)

// ---------- weight fp32[K,N] -> bf16 transposed Wt[N,K] ----------
__global__ __launch_bounds__(256) void wtrans_kernel(const float* __restrict__ W,
                                                     unsigned short* __restrict__ Wt,
                                                     int K, int N) {
  __shared__ float tile[32][33];
  const int n0 = blockIdx.x * 32, k0 = blockIdx.y * 32;
  const int tx = threadIdx.x & 31, ty = threadIdx.x >> 5;  // 32 x 8
#pragma unroll
  for (int i = 0; i < 4; ++i)
    tile[ty + 8 * i][tx] = W[(size_t)(k0 + ty + 8 * i) * N + n0 + tx];
  __syncthreads();
#pragma unroll
  for (int i = 0; i < 4; ++i)
    Wt[(size_t)(n0 + ty + 8 * i) * K + k0 + tx] = f2bf(tile[tx][ty + 8 * i]);
}

// ---------- LayerNorm: fp32 row -> bf16 normalized row ----------
__global__ __launch_bounds__(256) void ln_kernel(const float* __restrict__ x,
                                                 const float* __restrict__ g,
                                                 const float* __restrict__ b,
                                                 unsigned short* __restrict__ out) {
  __shared__ float shs[8], shs2[8];
  const size_t row = blockIdx.x;
  const float* xr = x + row * DM;
  float v[4], s = 0.f, s2 = 0.f;
#pragma unroll
  for (int i = 0; i < 4; ++i) {
    v[i] = xr[threadIdx.x + 256 * i];
    s += v[i];
    s2 += v[i] * v[i];
  }
#pragma unroll
  for (int off = 16; off >= 1; off >>= 1) {
    s  += __shfl_xor(s, off, 32);
    s2 += __shfl_xor(s2, off, 32);
  }
  const int wave = threadIdx.x >> 5, lane = threadIdx.x & 31;
  if (lane == 0) { shs[wave] = s; shs2[wave] = s2; }
  __syncthreads();
  float ts = 0.f, ts2 = 0.f;
#pragma unroll
  for (int w = 0; w < 8; ++w) { ts += shs[w]; ts2 += shs2[w]; }
  const float mu = ts * (1.0f / DM);
  const float var = ts2 * (1.0f / DM) - mu * mu;
  const float rs = rsqrtf(var + 1e-5f);
#pragma unroll
  for (int i = 0; i < 4; ++i) {
    const int c = threadIdx.x + 256 * i;
    out[row * DM + c] = f2bf((v[i] - mu) * rs * g[c] + b[c]);
  }
}

// ---------- WMMA GEMM: C = A[M,K](bf16) * Bt[N,K](bf16)^T ----------
// 128x128 block tile, 8 waves (4 over M x 2 over N, 32x64 each), K-step 32.
// A/B tiles staged in LDS via async DMA, double buffered.
struct EpiParams {
  const float* bias;          // [N]
  const float* resid;         // EPI_RESID
  float* outf;                // EPI_RESID
  unsigned short* outb;       // EPI_GELU
  unsigned short* q;          // EPI_QKV: [B,H,S,Hd]
  unsigned short* k;          // EPI_QKV: [B,H,S,Hd]
  unsigned short* vt;         // EPI_QKV: [B,H,Hd,S]
};
#define EPI_QKV   0
#define EPI_RESID 1
#define EPI_GELU  2

template <int EPI>
__global__ __launch_bounds__(256) void gemm_bf16_kernel(const unsigned short* __restrict__ A,
                                                        const unsigned short* __restrict__ Bt,
                                                        int M, int N, int K, EpiParams p) {
  // [buf][A/B][tile]
  __shared__ __align__(16) unsigned short sm[2][2][THALF];

  const int t = threadIdx.x;
  const int lane = t & 31;
  const int wave = t >> 5;
  const int lm = lane & 15, lh = lane >> 4;
  const int wm = (wave & 3) * 32;   // wave M offset in tile
  const int wn = (wave >> 2) * 64;  // wave N offset in tile
  const int m0 = blockIdx.x * 128;
  const int n0 = blockIdx.y * 128;

  // ---- async copy setup: each thread moves 2x16B for A and 2x16B for B per K-step
  const int crow = t >> 2;              // 0..63
  const int csub = (t & 3) * 8;         // element sub-column 0/8/16/24
  const unsigned smBase = (unsigned)(uintptr_t)&sm[0][0][0];
  const unsigned ldsA = smBase + (unsigned)(crow * TROW + csub) * 2;
  const unsigned ldsB = ldsA + TBYTES;
  unsigned voffA0 = ((unsigned)(m0 + crow) * (unsigned)K + csub) * 2;
  unsigned voffA1 = voffA0 + 64u * (unsigned)K * 2;
  unsigned voffB0 = ((unsigned)(n0 + crow) * (unsigned)K + csub) * 2;
  unsigned voffB1 = voffB0 + 64u * (unsigned)K * 2;

  auto issueTile = [&](int buf) {
    const unsigned bo = (unsigned)buf * (2 * TBYTES);
    ASYNC_CP_B128(ldsA + bo,          voffA0, A);
    ASYNC_CP_B128(ldsA + bo + CHUNK1, voffA1, A);
    ASYNC_CP_B128(ldsB + bo,          voffB0, Bt);
    ASYNC_CP_B128(ldsB + bo + CHUNK1, voffB1, Bt);
    voffA0 += 64; voffA1 += 64; voffB0 += 64; voffB1 += 64;  // K advances 32 halfs
  };

  const v8f vz = {0.f, 0.f, 0.f, 0.f, 0.f, 0.f, 0.f, 0.f};
  v8f acc[2][4];
#pragma unroll
  for (int mt = 0; mt < 2; ++mt)
#pragma unroll
    for (int nt = 0; nt < 4; ++nt) acc[mt][nt] = vz;

  const int NK = K >> 5;

  issueTile(0);
  WAIT_ASYNC0();
  __syncthreads();

  for (int kb = 0; kb < NK; ++kb) {
    const int cur = kb & 1;
    if (kb + 1 < NK) issueTile(cur ^ 1);   // overlap next tile's DMA with compute

    const unsigned short* As = &sm[cur][0][0];
    const unsigned short* Bs = &sm[cur][1][0];

    FragBF a[2], b[4];
#pragma unroll
    for (int mt = 0; mt < 2; ++mt) {
      const unsigned short* ap = As + (wm + mt * 16 + lm) * TROW + lh * 8;
      a[mt].u[0] = *(const v4u*)(ap);
      a[mt].u[1] = *(const v4u*)(ap + 16);
    }
#pragma unroll
    for (int nt = 0; nt < 4; ++nt) {
      const unsigned short* bp = Bs + (wn + nt * 16 + lm) * TROW + lh * 8;
      b[nt].u[0] = *(const v4u*)(bp);
      b[nt].u[1] = *(const v4u*)(bp + 16);
    }
#pragma unroll
    for (int mt = 0; mt < 2; ++mt)
#pragma unroll
      for (int nt = 0; nt < 4; ++nt)
        acc[mt][nt] = WMMA_BF16(a[mt].v, b[nt].v, acc[mt][nt]);

    WAIT_ASYNC0();      // our next-tile DMAs landed
    __syncthreads();    // everyone done reading cur + everyone's DMAs visible
  }

  // Epilogue. C layout: VGPR r -> row r + 8*lh, col = lm.
#pragma unroll
  for (int mt = 0; mt < 2; ++mt) {
#pragma unroll
    for (int nt = 0; nt < 4; ++nt) {
#pragma unroll
      for (int r = 0; r < 8; ++r) {
        const int gr = m0 + wm + mt * 16 + r + 8 * lh;
        const int gc = n0 + wn + nt * 16 + lm;
        float val = acc[mt][nt][r] + p.bias[gc];
        if (EPI == EPI_QKV) {
          const int which = gc >> 10;        // 0=q 1=k 2=v
          const int rem = gc & 1023;
          const int head = rem >> 6;
          const int hd = rem & 63;
          const int bi = gr >> 11;           // row / SEQ
          const int s = gr & 2047;
          const unsigned short bv = f2bf(val);
          if (which == 0)
            p.q[((size_t)(bi * NH + head) * SEQ + s) * HD + hd] = bv;
          else if (which == 1)
            p.k[((size_t)(bi * NH + head) * SEQ + s) * HD + hd] = bv;
          else
            p.vt[((size_t)(bi * NH + head) * HD + hd) * SEQ + s] = bv;
        } else if (EPI == EPI_RESID) {
          p.outf[(size_t)gr * N + gc] = val + p.resid[(size_t)gr * N + gc];
        } else {  // EPI_GELU
          p.outb[(size_t)gr * N + gc] = f2bf(gelu_exact(val));
        }
      }
    }
  }
}

// ---------- causal flash attention: one wave per 16-query tile ----------
__global__ __launch_bounds__(128) void attn_kernel(const unsigned short* __restrict__ Q,
                                                   const unsigned short* __restrict__ Km,
                                                   const unsigned short* __restrict__ Vt,
                                                   unsigned short* __restrict__ O) {
  const int lane = threadIdx.x & 31, wave = threadIdx.x >> 5;
  const int lm = lane & 15, lh = lane >> 4;
  const int q0 = (blockIdx.x * 4 + wave) * 16;
  const int head = blockIdx.y, batch = blockIdx.z;
  const size_t bh = (size_t)batch * NH + head;
  const unsigned short* Qh = Q + bh * SEQ * HD;
  const unsigned short* Kh = Km + bh * SEQ * HD;
  const unsigned short* Vh = Vt + bh * HD * SEQ;

  __shared__ __align__(16) unsigned short pl[4][16 * 32];
  unsigned short* pw = pl[wave];

  // Q fragments (hd split into two K=32 chunks)
  FragBF qa[2];
#pragma unroll
  for (int c = 0; c < 2; ++c) {
    qa[c].u[0] = *(const v4u*)(Qh + (size_t)(q0 + lm) * HD + c * 32 + lh * 8);
    qa[c].u[1] = *(const v4u*)(Qh + (size_t)(q0 + lm) * HD + c * 32 + lh * 8 + 16);
  }

  const v8f vz = {0.f, 0.f, 0.f, 0.f, 0.f, 0.f, 0.f, 0.f};
  v8f o[4];
  float mrow[8], lrow[8];
#pragma unroll
  for (int nt = 0; nt < 4; ++nt) o[nt] = vz;
#pragma unroll
  for (int r = 0; r < 8; ++r) { mrow[r] = -__builtin_inff(); lrow[r] = 0.f; }

  for (int k0 = 0; k0 <= q0 + 15; k0 += 32) {
    // ---- scores: Q (16x64) * K^T tile (64 x 32) -> two 16x16 tiles
    v8f st[2] = {vz, vz};
#pragma unroll
    for (int c = 0; c < 2; ++c) {
#pragma unroll
      for (int ct = 0; ct < 2; ++ct) {
        FragBF kb;  // B frag from K rows (contiguous hd per lane)
        kb.u[0] = *(const v4u*)(Kh + (size_t)(k0 + ct * 16 + lm) * HD + c * 32 + lh * 8);
        kb.u[1] = *(const v4u*)(Kh + (size_t)(k0 + ct * 16 + lm) * HD + c * 32 + lh * 8 + 16);
        st[ct] = WMMA_BF16(qa[c].v, kb.v, st[ct]);
      }
    }
    // ---- scale, causal mask, online softmax (rows reduced across 16-lane half-waves)
#pragma unroll
    for (int r = 0; r < 8; ++r) {
      const int row = q0 + r + 8 * lh;
      float s0 = st[0][r] * 0.125f;
      float s1 = st[1][r] * 0.125f;
      if (k0 + lm > row) s0 = -__builtin_inff();
      if (k0 + 16 + lm > row) s1 = -__builtin_inff();
      float mx = fmaxf(s0, s1);
#pragma unroll
      for (int off = 1; off <= 8; off <<= 1) mx = fmaxf(mx, __shfl_xor(mx, off, 32));
      const float mn = fmaxf(mrow[r], mx);
      const float cr = __expf(mrow[r] - mn);
      mrow[r] = mn;
      const float p0 = __expf(s0 - mn);
      const float p1 = __expf(s1 - mn);
      float sum = p0 + p1;
#pragma unroll
      for (int off = 1; off <= 8; off <<= 1) sum += __shfl_xor(sum, off, 32);
      lrow[r] = lrow[r] * cr + sum;
#pragma unroll
      for (int nt = 0; nt < 4; ++nt) o[nt][r] *= cr;
      pw[(r + 8 * lh) * 32 + lm] = f2bf(p0);
      pw[(r + 8 * lh) * 32 + 16 + lm] = f2bf(p1);
    }
    // wave-local LDS ordering: stores above must land before A-frag reload below
    asm volatile("s_wait_dscnt 0" ::: "memory");
    FragBF pa;  // P as 16x32 A fragment
    pa.u[0] = *(const v4u*)(pw + lm * 32 + lh * 8);
    pa.u[1] = *(const v4u*)(pw + lm * 32 + lh * 8 + 16);
    // ---- O += P (16x32) * V tile (32 x 64), B frags from V^T (contiguous keys per lane)
#pragma unroll
    for (int nt = 0; nt < 4; ++nt) {
      FragBF vb;
      vb.u[0] = *(const v4u*)(Vh + (size_t)(nt * 16 + lm) * SEQ + k0 + lh * 8);
      vb.u[1] = *(const v4u*)(Vh + (size_t)(nt * 16 + lm) * SEQ + k0 + lh * 8 + 16);
      o[nt] = WMMA_BF16(pa.v, vb.v, o[nt]);
    }
  }
  // ---- write attn_out bf16, layout [B,S,H*Hd] (heads concatenated)
#pragma unroll
  for (int nt = 0; nt < 4; ++nt) {
#pragma unroll
    for (int r = 0; r < 8; ++r) {
      const int row = q0 + r + 8 * lh;
      const float val = o[nt][r] / lrow[r];
      O[((size_t)batch * SEQ + row) * DM + head * HD + nt * 16 + lm] = f2bf(val);
    }
  }
}

// ---------- host-side orchestration ----------
extern "C" void kernel_launch(void* const* d_in, const int* in_sizes, int n_in,
                              void* d_out, int out_size, void* d_ws, size_t ws_size,
                              hipStream_t stream) {
  (void)in_sizes; (void)n_in; (void)out_size; (void)ws_size;
  const float* x      = (const float*)d_in[0];
  const float* qkv_w  = (const float*)d_in[1];
  const float* qkv_b  = (const float*)d_in[2];
  const float* out_w  = (const float*)d_in[3];
  const float* out_b  = (const float*)d_in[4];
  const float* ffn1_w = (const float*)d_in[5];
  const float* ffn1_b = (const float*)d_in[6];
  const float* ffn2_w = (const float*)d_in[7];
  const float* ffn2_b = (const float*)d_in[8];
  const float* ln1_g  = (const float*)d_in[9];
  const float* ln1_b  = (const float*)d_in[10];
  const float* ln2_g  = (const float*)d_in[11];
  const float* ln2_b  = (const float*)d_in[12];
  float* out = (float*)d_out;

  char* ws = (char*)d_ws;
  size_t off = 0;
  auto alloc = [&](size_t bytes) -> char* {
    char* p = ws + off;
    off += (bytes + 255) & ~(size_t)255;
    return p;
  };
  unsigned short* wt_qkv = (unsigned short*)alloc((size_t)DM * 3 * DM * 2);
  unsigned short* wt_out = (unsigned short*)alloc((size_t)DM * DM * 2);
  unsigned short* wt_f1  = (unsigned short*)alloc((size_t)DM * DFF * 2);
  unsigned short* wt_f2  = (unsigned short*)alloc((size_t)DFF * DM * 2);
  unsigned short* normed = (unsigned short*)alloc((size_t)MROWS * DM * 2);
  unsigned short* qbuf   = (unsigned short*)alloc((size_t)MROWS * DM * 2);
  unsigned short* kbuf   = (unsigned short*)alloc((size_t)MROWS * DM * 2);
  unsigned short* vtbuf  = (unsigned short*)alloc((size_t)MROWS * DM * 2);
  unsigned short* attnb  = (unsigned short*)alloc((size_t)MROWS * DM * 2);
  float*          x1     = (float*)alloc((size_t)MROWS * DM * 4);
  unsigned short* h2n    = (unsigned short*)alloc((size_t)MROWS * DM * 2);
  unsigned short* f1act  = (unsigned short*)alloc((size_t)MROWS * DFF * 2);

  // 1) weights -> bf16 transposed [N,K]
  wtrans_kernel<<<dim3(3 * DM / 32, DM / 32), 256, 0, stream>>>(qkv_w, wt_qkv, DM, 3 * DM);
  wtrans_kernel<<<dim3(DM / 32, DM / 32), 256, 0, stream>>>(out_w, wt_out, DM, DM);
  wtrans_kernel<<<dim3(DFF / 32, DM / 32), 256, 0, stream>>>(ffn1_w, wt_f1, DM, DFF);
  wtrans_kernel<<<dim3(DM / 32, DFF / 32), 256, 0, stream>>>(ffn2_w, wt_f2, DFF, DM);

  // 2) LN1 -> bf16
  ln_kernel<<<MROWS, 256, 0, stream>>>(x, ln1_g, ln1_b, normed);

  // 3) QKV projection (split epilogue, V stored transposed)
  EpiParams pq = {qkv_b, nullptr, nullptr, nullptr, qbuf, kbuf, vtbuf};
  gemm_bf16_kernel<EPI_QKV><<<dim3(MROWS / 128, 3 * DM / 128), 256, 0, stream>>>(
      normed, wt_qkv, MROWS, 3 * DM, DM, pq);

  // 4) causal attention
  attn_kernel<<<dim3(SEQ / 16 / 4, NH, BATCH), 128, 0, stream>>>(qbuf, kbuf, vtbuf, attnb);

  // 5) output projection + residual -> x1 (f32)
  EpiParams po = {out_b, x, x1, nullptr, nullptr, nullptr, nullptr};
  gemm_bf16_kernel<EPI_RESID><<<dim3(MROWS / 128, DM / 128), 256, 0, stream>>>(
      attnb, wt_out, MROWS, DM, DM, po);

  // 6) LN2 -> bf16
  ln_kernel<<<MROWS, 256, 0, stream>>>(x1, ln2_g, ln2_b, h2n);

  // 7) FFN1 + exact GELU -> bf16
  EpiParams pf1 = {ffn1_b, nullptr, nullptr, f1act, nullptr, nullptr, nullptr};
  gemm_bf16_kernel<EPI_GELU><<<dim3(MROWS / 128, DFF / 128), 256, 0, stream>>>(
      h2n, wt_f1, MROWS, DFF, DM, pf1);

  // 8) FFN2 + residual(x1) -> d_out (f32)
  EpiParams pf2 = {ffn2_b, x1, out, nullptr, nullptr, nullptr, nullptr};
  gemm_bf16_kernel<EPI_RESID><<<dim3(MROWS / 128, DM / 128), 256, 0, stream>>>(
      f1act, wt_f2, MROWS, DM, DFF, pf2);
}
